// VanillaGNN_5858335391843
// MI455X (gfx1250) — matrix operand order repaired and checked by reading, not compiled, not run on hardware
//
#include <hip/hip_runtime.h>
#include <hip/hip_bf16.h>
#include <stddef.h>

#define N_NODES 50000            // == 3125 * 16, no row guards needed
#define N_EDGES 800000
#define NFEAT   128
#define NHID    96
#define NCLASS  40
#define NCLASS_PAD 48
#define NLAYERS 3

typedef __attribute__((ext_vector_type(16))) __bf16 v16bf;
typedef __attribute__((ext_vector_type(8)))  __bf16 v8bf;
typedef __attribute__((ext_vector_type(8)))  float  v8f;

// ---------------------------------------------------------------------------
// WMMA fragment helpers (CDNA5 wave32 layouts, CDNA5 ISA §7.12.2)
// A row-major bf16 [M x LDA]: lane fragment = two contiguous 16B chunks.
//   lanes 0-15 (row m0+l):  halves 0-7 = K k0..k0+7, halves 8-15 = K k0+16..k0+23
//   lanes 16-31:            K base shifted by +8
// ---------------------------------------------------------------------------
__device__ __forceinline__ v16bf load_a_frag(const __bf16* __restrict__ A, int ldA,
                                             int m0, int k0, int lane) {
  int m  = m0 + (lane & 15);
  int kb = k0 + ((lane & 16) ? 8 : 0);
  const __bf16* p = A + (size_t)m * ldA + kb;
  v8bf lo = *(const v8bf*)(p);        // K = kb .. kb+7      (global_load_b128)
  v8bf hi = *(const v8bf*)(p + 16);   // K = kb+16 .. kb+23  (global_load_b128)
  v16bf a;
#pragma unroll
  for (int i = 0; i < 8; ++i) { a[i] = lo[i]; a[8 + i] = hi[i]; }
  return a;
}

// B 32x16 (KxN) from B^T stored row-major [N x K] bf16: same K order as A frag.
// lanes 0-15: col n0+l, K k0..k0+7, k0+16..k0+23 ; lanes 16-31: K base shifted by +8.
__device__ __forceinline__ v16bf load_b_frag(const __bf16* __restrict__ BT, int ldK,
                                             int n0, int k0, int lane) {
  int n  = n0 + (lane & 15);
  int kb = k0 + ((lane & 16) ? 8 : 0);
  const __bf16* p = BT + (size_t)n * ldK + kb;
  v8bf lo = *(const v8bf*)(p);
  v8bf hi = *(const v8bf*)(p + 16);
  v16bf b;
#pragma unroll
  for (int i = 0; i < 8; ++i) { b[i] = lo[i]; b[8 + i] = hi[i]; }
  return b;
}

// ---------------------------------------------------------------------------
// Templated GEMM: one 16x16 output tile per wave32, compile-time K/N/ld.
// C/D layout: lanes 0-15 -> N=lane, rows r=0..7; lanes 16-31 -> rows 8+r.
// ---------------------------------------------------------------------------
template <int K, int NC, int LDA, int LDC, bool RELU, bool BF16OUT>
__global__ __launch_bounds__(256) void gemm_tile(const __bf16* __restrict__ A,
                                                 const __bf16* __restrict__ BT,
                                                 const float* __restrict__ bias,
                                                 void* __restrict__ Cout) {
  constexpr int TILES_M = N_NODES / 16;
  constexpr int TILES_N = (NC + 15) / 16;
  int wave = threadIdx.x >> 5;
  int lane = threadIdx.x & 31;
  int tile = blockIdx.x * (blockDim.x >> 5) + wave;
  if (tile >= TILES_M * TILES_N) return;       // wave-uniform
  int mt = tile / TILES_N, nt = tile - mt * TILES_N;
  int m0 = mt << 4, n0 = nt << 4;

  v8f acc = {};
#pragma unroll
  for (int k0 = 0; k0 < K; k0 += 32) {
    v16bf a = load_a_frag(A, LDA, m0, k0, lane);
    v16bf b = load_b_frag(BT, K, n0, k0, lane);
    acc = __builtin_amdgcn_wmma_f32_16x16x32_bf16(false, a, false, b,
                                                  (short)0, acc, false, false);
  }

  int n  = n0 + (lane & 15);
  int mb = m0 + ((lane & 16) ? 8 : 0);
  if ((NC & 15) != 0 && n >= NC) return;       // decoder col guard (post-WMMA)
  float bv = bias[n];
  if constexpr (BF16OUT) {
    __bf16* C = (__bf16*)Cout;
#pragma unroll
    for (int r = 0; r < 8; ++r) {
      float v = acc[r] + bv;
      if (RELU) v = fmaxf(v, 0.0f);
      C[(size_t)(mb + r) * LDC + n] = (__bf16)v;
    }
  } else {
    float* C = (float*)Cout;
#pragma unroll
    for (int r = 0; r < 8; ++r) {
      float v = acc[r] + bv;
      if (RELU) v = fmaxf(v, 0.0f);
      C[(size_t)(mb + r) * LDC + n] = v;
    }
  }
}

// Fused SAGE combine: out = relu(mean @ w_l + h @ w_r + b_l), all NHID, bf16 out
__global__ __launch_bounds__(256) void sage_combine(const __bf16* __restrict__ mean,
                                                    const __bf16* __restrict__ wlT,
                                                    const __bf16* __restrict__ h,
                                                    const __bf16* __restrict__ wrT,
                                                    const float* __restrict__ bias,
                                                    __bf16* __restrict__ C) {
  constexpr int TILES_M = N_NODES / 16;
  constexpr int TILES_N = NHID / 16;           // 6
  int wave = threadIdx.x >> 5;
  int lane = threadIdx.x & 31;
  int tile = blockIdx.x * (blockDim.x >> 5) + wave;
  if (tile >= TILES_M * TILES_N) return;
  int mt = tile / TILES_N, nt = tile - mt * TILES_N;
  int m0 = mt << 4, n0 = nt << 4;

  v8f acc = {};
#pragma unroll
  for (int k0 = 0; k0 < NHID; k0 += 32) {      // mean @ w_l
    v16bf a = load_a_frag(mean, NHID, m0, k0, lane);
    v16bf b = load_b_frag(wlT, NHID, n0, k0, lane);
    acc = __builtin_amdgcn_wmma_f32_16x16x32_bf16(false, a, false, b,
                                                  (short)0, acc, false, false);
  }
#pragma unroll
  for (int k0 = 0; k0 < NHID; k0 += 32) {      // + h @ w_r
    v16bf a = load_a_frag(h, NHID, m0, k0, lane);
    v16bf b = load_b_frag(wrT, NHID, n0, k0, lane);
    acc = __builtin_amdgcn_wmma_f32_16x16x32_bf16(false, a, false, b,
                                                  (short)0, acc, false, false);
  }

  int n  = n0 + (lane & 15);
  int mb = m0 + ((lane & 16) ? 8 : 0);
  float bv = bias[n];
#pragma unroll
  for (int r = 0; r < 8; ++r) {
    float v = fmaxf(acc[r] + bv, 0.0f);
    C[(size_t)(mb + r) * NHID + n] = (__bf16)v;
  }
}

// ---------------------------------------------------------------------------
// Graph aggregation + utility kernels
// ---------------------------------------------------------------------------
__global__ void zero_f32(float* __restrict__ p, size_t n) {
  size_t i = (size_t)blockIdx.x * blockDim.x + threadIdx.x;
  if (i < n) p[i] = 0.0f;
}

__global__ void zero_bf16(__bf16* __restrict__ p, int n) {
  int i = blockIdx.x * blockDim.x + threadIdx.x;
  if (i < n) p[i] = (__bf16)0.0f;
}

__global__ void cvt_bf16_kernel(const float* __restrict__ in, __bf16* __restrict__ out, int n) {
  int i = blockIdx.x * blockDim.x + threadIdx.x;
  if (i < n) out[i] = (__bf16)in[i];
}

// BT[n*K + k] = (bf16)W[k*Nc + n]   (W is K x Nc row-major f32)
__global__ void transpose_bf16_kernel(const float* __restrict__ W, __bf16* __restrict__ BT,
                                      int K, int Nc) {
  int i = blockIdx.x * blockDim.x + threadIdx.x;
  if (i >= K * Nc) return;
  int k = i / Nc, n = i - k * Nc;
  BT[(size_t)n * K + k] = (__bf16)W[i];
}

__global__ void degree_kernel(const int* __restrict__ dst, float* __restrict__ cnt, int E) {
  int e = blockIdx.x * blockDim.x + threadIdx.x;
  if (e < E) atomicAdd(&cnt[dst[e]], 1.0f);
}

// gather bf16 features of src node, f32 atomic scatter-add to dst row
__global__ void scatter_kernel(const __bf16* __restrict__ h, const int* __restrict__ src,
                               const int* __restrict__ dst, float* __restrict__ ssum, int E) {
  int e = blockIdx.x * blockDim.x + threadIdx.x;
  if (e >= E) return;
  size_t s = (size_t)src[e] * NHID;
  size_t d = (size_t)dst[e] * NHID;
#pragma unroll
  for (int c = 0; c < NHID / 8; ++c) {
    v8bf v = *(const v8bf*)(h + s + 8 * c);   // 16B vector gather
#pragma unroll
    for (int i = 0; i < 8; ++i) {
      atomicAdd(&ssum[d + 8 * c + i], (float)v[i]);
    }
  }
}

// mean (f32) -> bf16 feature matrix for the WMMA combine
__global__ void mean_to_bf16(const float* __restrict__ ssum, const float* __restrict__ cnt,
                             __bf16* __restrict__ meanbf, size_t n) {
  size_t i = (size_t)blockIdx.x * blockDim.x + threadIdx.x;
  if (i >= n) return;
  float c = cnt[i / NHID];
  meanbf[i] = (__bf16)((c > 0.0f) ? (ssum[i] / c) : 0.0f);
}

// ---------------------------------------------------------------------------
// Host side
// ---------------------------------------------------------------------------
static inline size_t align256(size_t x) { return (x + 255) & ~(size_t)255; }

extern "C" void kernel_launch(void* const* d_in, const int* in_sizes, int n_in,
                              void* d_out, int out_size, void* d_ws, size_t ws_size,
                              hipStream_t stream) {
  (void)in_sizes; (void)n_in; (void)out_size; (void)ws_size;

  const float* x       = (const float*)d_in[0];
  const int*   ei      = (const int*)d_in[1];
  const float* enc_w   = (const float*)d_in[2];
  const float* enc_b   = (const float*)d_in[3];
  const float* lin_l_w = (const float*)d_in[4];
  const float* lin_l_b = (const float*)d_in[5];
  const float* lin_r_w = (const float*)d_in[6];
  const float* dec_w   = (const float*)d_in[7];
  const float* dec_b   = (const float*)d_in[8];
  float* out = (float*)d_out;

  const int* src = ei;
  const int* dst = ei + N_EDGES;

  // Workspace carve-out (all bf16 regions 256B-aligned -> 16/32B frag loads OK)
  char* ws = (char*)d_ws;
  size_t off = 0;
  __bf16* xbf    = (__bf16*)(ws + off); off = align256(off + (size_t)N_NODES * NFEAT * sizeof(__bf16));
  __bf16* hA     = (__bf16*)(ws + off); off = align256(off + (size_t)N_NODES * NHID * sizeof(__bf16));
  __bf16* hB     = (__bf16*)(ws + off); off = align256(off + (size_t)N_NODES * NHID * sizeof(__bf16));
  __bf16* meanbf = (__bf16*)(ws + off); off = align256(off + (size_t)N_NODES * NHID * sizeof(__bf16));
  float*  ssum   = (float*)(ws + off);  off = align256(off + (size_t)N_NODES * NHID * sizeof(float));
  float*  cnt    = (float*)(ws + off);  off = align256(off + (size_t)N_NODES * sizeof(float));
  __bf16* encBT  = (__bf16*)(ws + off); off = align256(off + (size_t)NHID * NFEAT * sizeof(__bf16));
  __bf16* wlBT   = (__bf16*)(ws + off); off = align256(off + (size_t)NLAYERS * NHID * NHID * sizeof(__bf16));
  __bf16* wrBT   = (__bf16*)(ws + off); off = align256(off + (size_t)NLAYERS * NHID * NHID * sizeof(__bf16));
  __bf16* decBT  = (__bf16*)(ws + off); off = align256(off + (size_t)NCLASS_PAD * NHID * sizeof(__bf16));

  const int B = 256;

  // Activations / weights -> bf16 (weights transposed to [N x K])
  cvt_bf16_kernel<<<(N_NODES * NFEAT + B - 1) / B, B, 0, stream>>>(x, xbf, N_NODES * NFEAT);
  transpose_bf16_kernel<<<(NFEAT * NHID + B - 1) / B, B, 0, stream>>>(enc_w, encBT, NFEAT, NHID);
  for (int l = 0; l < NLAYERS; ++l) {
    transpose_bf16_kernel<<<(NHID * NHID + B - 1) / B, B, 0, stream>>>(
        lin_l_w + (size_t)l * NHID * NHID, wlBT + (size_t)l * NHID * NHID, NHID, NHID);
    transpose_bf16_kernel<<<(NHID * NHID + B - 1) / B, B, 0, stream>>>(
        lin_r_w + (size_t)l * NHID * NHID, wrBT + (size_t)l * NHID * NHID, NHID, NHID);
  }
  zero_bf16<<<(NCLASS_PAD * NHID + B - 1) / B, B, 0, stream>>>(decBT, NCLASS_PAD * NHID);
  transpose_bf16_kernel<<<(NHID * NCLASS + B - 1) / B, B, 0, stream>>>(dec_w, decBT, NHID, NCLASS);

  // Degrees (dst list is layer-invariant)
  zero_f32<<<(N_NODES + B - 1) / B, B, 0, stream>>>(cnt, (size_t)N_NODES);
  degree_kernel<<<(N_EDGES + B - 1) / B, B, 0, stream>>>(dst, cnt, N_EDGES);

  constexpr int TILES_M = N_NODES / 16;               // 3125
  constexpr int WPB = 256 / 32;                       // 8 waves / block
  const int hid_blocks = (TILES_M * (NHID / 16) + WPB - 1) / WPB;
  const int dec_blocks = (TILES_M * ((NCLASS + 15) / 16) + WPB - 1) / WPB;

  // Encoder: hA = relu(x @ enc_w + enc_b)   [bf16 out]
  gemm_tile<NFEAT, NHID, NFEAT, NHID, true, true>
      <<<hid_blocks, B, 0, stream>>>(xbf, encBT, enc_b, hA);

  const size_t feat_elems = (size_t)N_NODES * NHID;
  __bf16* cur = hA;
  __bf16* nxt = hB;
  for (int layer = 0; layer < NLAYERS; ++layer) {
    zero_f32<<<(unsigned)((feat_elems + B - 1) / B), B, 0, stream>>>(ssum, feat_elems);
    scatter_kernel<<<(N_EDGES + B - 1) / B, B, 0, stream>>>(cur, src, dst, ssum, N_EDGES);
    mean_to_bf16<<<(unsigned)((feat_elems + B - 1) / B), B, 0, stream>>>(ssum, cnt, meanbf, feat_elems);
    sage_combine<<<hid_blocks, B, 0, stream>>>(
        meanbf, wlBT + (size_t)layer * NHID * NHID,
        cur,    wrBT + (size_t)layer * NHID * NHID,
        lin_l_b + (size_t)layer * NHID, nxt);
    __bf16* t = cur; cur = nxt; nxt = t;
  }

  // Decoder: out = cur @ dec_w + dec_b  (f32 out, col-guarded store for N=40)
  gemm_tile<NHID, NCLASS, NHID, NCLASS, false, false>
      <<<dec_blocks, B, 0, stream>>>(cur, decBT, dec_b, out);
}
